// PMPGNN_85641647882793
// MI455X (gfx1250) — matrix operand order repaired and checked by reading, not compiled, not run on hardware
//
#include <hip/hip_runtime.h>
#include <hip/hip_bf16.h>

#define NN      100000
#define EE      1600000
#define NFEAT_  512
#define NHID_   256
#define NCLS_   64
#define KITER_  8
#define ALPHA_  0.1f
#define NSLOPE_ 0.2f

typedef __attribute__((ext_vector_type(16))) __bf16 v16bf;
typedef __attribute__((ext_vector_type(8)))  float  v8f;
typedef __attribute__((ext_vector_type(4)))  unsigned int v4u;

union V16U { v16bf v; v4u q[2]; };

__device__ __forceinline__ v8f wmma_bf16(v16bf a, v16bf b, v8f c) {
  return __builtin_amdgcn_wmma_f32_16x16x32_bf16(false, a, false, b, (short)0, c,
                                                 false, false);
}

__device__ __forceinline__ float eluf(float x) { return x > 0.f ? x : expm1f(x); }

// 32B fragment load as two b128 (works for LDS and global pointers)
__device__ __forceinline__ v16bf frag_load(const __bf16* q) {
  V16U t;
  t.q[0] = *(const v4u*)q;
  t.q[1] = *(const v4u*)(q + 8);
  return t.v;
}

// ---------------------------------------------------------------------------
// Pack a row-major KxN fp32 weight matrix into WMMA B-fragment order, bf16.
// Fragment (kt,nt): lane half h holds K = kt*32 + h*16 + e, N = nt*16 + (lane&15).
// Storage: ((kt*NT + nt)*32 + lane)*16 + e   (contiguous 32B per lane;
// one kt-panel of W_in = 16 fragments = 16KB contiguous).
// ---------------------------------------------------------------------------
__global__ void frag_convert_kernel(const float* __restrict__ W,
                                    __bf16* __restrict__ out, int K, int Ncols) {
  int tid = blockIdx.x * blockDim.x + threadIdx.x;
  if (tid >= K * Ncols) return;
  int NT   = Ncols >> 4;
  int e    = tid & 15;
  int lane = (tid >> 4) & 31;
  int f    = tid >> 9;
  int kt = f / NT, nt = f % NT;
  int kk = kt * 32 + ((lane >> 4) << 4) + e;
  int nn = nt * 16 + (lane & 15);
  out[tid] = (__bf16)W[kk * Ncols + nn];
}

// ---------------------------------------------------------------------------
// h = elu(x @ W_in) @ W_out   (fused, one wave per 16-row tile, 4 waves/block)
// Stage 1: W_in k-panel staged in LDS (shared by the 4 waves); next panel's
// global loads are issued before the wmma chain and ds_stored after it, so
// L2 latency hides under 16 wmmas. B fragments are register double-buffered
// from LDS: ds_loads for fragment nt+1 are issued before the wmma of nt.
// ---------------------------------------------------------------------------
__global__ __launch_bounds__(128) void gemm_h_kernel(
    const float* __restrict__ x, const __bf16* __restrict__ winf,
    const __bf16* __restrict__ woutf, float* __restrict__ hbuf) {
  __shared__ __bf16 bpan[16 * 32 * 16];      // 16KB: one W_in kt-panel
  __shared__ __bf16 stage[4 * 16 * NHID_];   // 32KB: elu handoff
  const int tid  = threadIdx.x;
  const int lane = tid & 31;
  const int wib  = tid >> 5;
  const int gw   = blockIdx.x * 4 + wib;
  const bool active = gw < (NN / 16);
  const int rowBase = gw * 16;
  const int m  = lane & 15;
  const int hh = lane >> 4;
  v8f zero = {};
  v8f acc[16];
#pragma unroll
  for (int i = 0; i < 16; ++i) acc[i] = zero;

  // prologue: cooperative copy of panel 0 (128B per thread)
  {
    const v4u* g = (const v4u*)winf;
    v4u* l = (v4u*)bpan;
#pragma unroll
    for (int c = 0; c < 8; ++c) l[tid * 8 + c] = g[tid * 8 + c];
  }
  __syncthreads();

  const float* arow = x + (size_t)(rowBase + m) * NFEAT_ + hh * 16;
  v16bf a;
  if (active) {
    const float* p = arow;
#pragma unroll
    for (int e = 0; e < 8; ++e) a[e] = (__bf16)p[e];
#pragma unroll
    for (int e = 0; e < 8; ++e) a[8 + e] = (__bf16)p[8 + e];
  }

  for (int kt = 0; kt < NFEAT_ / 32; ++kt) {
    const bool hn = (kt + 1 < NFEAT_ / 32);
    // issue next-panel global loads up front; held in regs across the chain
    v4u tmp[8];
    if (hn) {
      const v4u* g = (const v4u*)(winf + (size_t)(kt + 1) * (16 * 32 * 16));
#pragma unroll
      for (int c = 0; c < 8; ++c) tmp[c] = g[tid * 8 + c];
    }
    float an[16];
    if (active) {
      if (hn) {
        const float* p = arow + (kt + 1) * 32;
#pragma unroll
        for (int e = 0; e < 8; ++e) an[e] = p[e];
#pragma unroll
        for (int e = 0; e < 8; ++e) an[8 + e] = p[8 + e];
      }
      __builtin_prefetch(arow + (kt + 2) * 32, 0, 3);
      // wmma chain: B fragments from LDS, register double-buffered so the
      // ds_loads of fragment nt+1 are in flight during the wmma of nt.
      v16bf bcur = frag_load(bpan + lane * 16);
#pragma unroll
      for (int nt = 0; nt < 16; ++nt) {
        v16bf bnext = bcur;
        if (nt + 1 < 16) bnext = frag_load(bpan + ((nt + 1) * 32 + lane) * 16);
        acc[nt] = wmma_bf16(a, bcur, acc[nt]);
        bcur = bnext;
      }
#if __has_builtin(__builtin_amdgcn_sched_group_barrier)
      __builtin_amdgcn_sched_group_barrier(0x100, 2, 0);    // frag 0 loads
#pragma unroll
      for (int i = 0; i < 15; ++i) {
        __builtin_amdgcn_sched_group_barrier(0x100, 2, 0);  // frag i+1 loads
        __builtin_amdgcn_sched_group_barrier(0x008, 1, 0);  // wmma i
      }
      __builtin_amdgcn_sched_group_barrier(0x008, 1, 0);    // wmma 15
#endif
    }
    __syncthreads();  // all waves done reading panel kt
    if (hn) {
      v4u* l = (v4u*)bpan;
#pragma unroll
      for (int c = 0; c < 8; ++c) l[tid * 8 + c] = tmp[c];
      if (active) {
#pragma unroll
        for (int e = 0; e < 16; ++e) a[e] = (__bf16)an[e];
      }
    }
    __syncthreads();  // panel kt+1 visible
  }

  // elu -> bf16 LDS staging (C layout: row = hh*8+r, col = nt*16+m)
  if (active) {
    __bf16* my = stage + wib * 16 * NHID_;
#pragma unroll
    for (int nt = 0; nt < 16; ++nt)
#pragma unroll
      for (int r = 0; r < 8; ++r)
        my[(hh * 8 + r) * NHID_ + nt * 16 + m] = (__bf16)eluf(acc[nt][r]);
  }
  __syncthreads();
  if (active) {
    const __bf16* myb = stage + wib * 16 * NHID_ + m * NHID_ + hh * 16;
    v8f c2[4];
#pragma unroll
    for (int i = 0; i < 4; ++i) c2[i] = zero;
    v16bf a2 = frag_load(myb);
    for (int kt = 0; kt < NHID_ / 32; ++kt) {
      const __bf16* bptr = woutf + ((size_t)kt * 4 * 32 + lane) * 16;
      v16bf b[4];
#pragma unroll
      for (int nt = 0; nt < 4; ++nt) b[nt] = frag_load(bptr + nt * 32 * 16);
#pragma unroll
      for (int nt = 0; nt < 4; ++nt) c2[nt] = wmma_bf16(a2, b[nt], c2[nt]);
      if (kt + 1 < NHID_ / 32) a2 = frag_load(myb + (kt + 1) * 32);
    }
#pragma unroll
    for (int nt = 0; nt < 4; ++nt)
#pragma unroll
      for (int r = 0; r < 8; ++r)
        hbuf[(size_t)(rowBase + hh * 8 + r) * NCLS_ + nt * 16 + m] = c2[nt][r];
  }
}

// ---------------------------------------------------------------------------
// hl = elu(elu(latp @ Ws1 + b1) @ Ws2 + b2)   (fused, one wave per 16 rows)
// ---------------------------------------------------------------------------
__global__ __launch_bounds__(128) void gemm_hl_kernel(
    const float* __restrict__ latp, const __bf16* __restrict__ ws1f,
    const __bf16* __restrict__ ws2f, const float* __restrict__ b1,
    const float* __restrict__ b2, float* __restrict__ hlbuf) {
  __shared__ __bf16 lds[4 * 16 * NCLS_];  // 8KB
  const int lane = threadIdx.x & 31;
  const int wib  = threadIdx.x >> 5;
  const int gw   = blockIdx.x * 4 + wib;
  const bool active = gw < (NN / 16);
  const int rowBase = gw * 16;
  const int m  = lane & 15;
  const int hh = lane >> 4;
  v8f zero = {};
  v8f acc[4];
#pragma unroll
  for (int i = 0; i < 4; ++i) acc[i] = zero;

  if (active) {
    const float* arow = latp + (size_t)(rowBase + m) * NCLS_ + hh * 16;
#pragma unroll
    for (int kt = 0; kt < 2; ++kt) {
      v16bf a;
      const float* p = arow + kt * 32;
#pragma unroll
      for (int e = 0; e < 8; ++e) a[e] = (__bf16)p[e];
#pragma unroll
      for (int e = 0; e < 8; ++e) a[8 + e] = (__bf16)p[8 + e];
      const __bf16* bptr = ws1f + ((size_t)kt * 4 * 32 + lane) * 16;
      v16bf b[4];
#pragma unroll
      for (int nt = 0; nt < 4; ++nt) b[nt] = frag_load(bptr + nt * 32 * 16);
#pragma unroll
      for (int nt = 0; nt < 4; ++nt) acc[nt] = wmma_bf16(a, b[nt], acc[nt]);
    }
    __bf16* my = lds + wib * 16 * NCLS_;
#pragma unroll
    for (int nt = 0; nt < 4; ++nt)
#pragma unroll
      for (int r = 0; r < 8; ++r)
        my[(hh * 8 + r) * NCLS_ + nt * 16 + m] =
            (__bf16)eluf(acc[nt][r] + b1[nt * 16 + m]);
  }
  __syncthreads();
  if (active) {
    const __bf16* myb = lds + wib * 16 * NCLS_ + m * NCLS_ + hh * 16;
    v8f c2[4];
#pragma unroll
    for (int i = 0; i < 4; ++i) c2[i] = zero;
#pragma unroll
    for (int kt = 0; kt < 2; ++kt) {
      v16bf a = frag_load(myb + kt * 32);
      const __bf16* bptr = ws2f + ((size_t)kt * 4 * 32 + lane) * 16;
      v16bf b[4];
#pragma unroll
      for (int nt = 0; nt < 4; ++nt) b[nt] = frag_load(bptr + nt * 32 * 16);
#pragma unroll
      for (int nt = 0; nt < 4; ++nt) c2[nt] = wmma_bf16(a, b[nt], c2[nt]);
    }
#pragma unroll
    for (int nt = 0; nt < 4; ++nt)
#pragma unroll
      for (int r = 0; r < 8; ++r)
        hlbuf[(size_t)(rowBase + hh * 8 + r) * NCLS_ + nt * 16 + m] =
            eluf(c2[nt][r] + b2[nt * 16 + m]);
  }
}

// ---------------------------------------------------------------------------
// el/er node attention scalars (one wave per node, 2 cols per lane)
// ---------------------------------------------------------------------------
__global__ void attn_kernel(const float* __restrict__ h,
                            const float* __restrict__ attn_l,
                            const float* __restrict__ attn_r,
                            float* __restrict__ el, float* __restrict__ er) {
  int gw   = (blockIdx.x * blockDim.x + threadIdx.x) >> 5;
  int lane = threadIdx.x & 31;
  if (gw >= NN) return;
  float h0 = h[(size_t)gw * 64 + lane];
  float h1 = h[(size_t)gw * 64 + 32 + lane];
  float l0 = h0 > 0.f ? h0 : NSLOPE_ * h0;
  float l1 = h1 > 0.f ? h1 : NSLOPE_ * h1;
  float pl = l0 * attn_l[lane] + l1 * attn_l[32 + lane];
  float pr = l0 * attn_r[lane] + l1 * attn_r[32 + lane];
#pragma unroll
  for (int s = 16; s >= 1; s >>= 1) {
    pl += __shfl_xor(pl, s);
    pr += __shfl_xor(pr, s);
  }
  if (lane == 0) { el[gw] = pl; er[gw] = pr; }
}

// ---------------------------------------------------------------------------
// Per-edge: ew = exp(e - betaw*d) + 1e-9, degree atomics, lp_loss partial.
// One wave per edge; h/hl gathers are L2-resident (51 MB << 192 MB L2).
// ---------------------------------------------------------------------------
__global__ void edge_kernel(const int* __restrict__ src, const int* __restrict__ dst,
                            const float* __restrict__ h, const float* __restrict__ hl,
                            const float* __restrict__ el, const float* __restrict__ er,
                            const float* __restrict__ s_attn,
                            const float* __restrict__ weights,
                            const float* __restrict__ beta, const float* __restrict__ aw,
                            float* __restrict__ ew, float* __restrict__ deg_s,
                            float* __restrict__ deg_d, float* __restrict__ lp) {
  __shared__ float lds_lp[8];
  int gw   = (blockIdx.x * blockDim.x + threadIdx.x) >> 5;
  int lane = threadIdx.x & 31;
  int wib  = threadIdx.x >> 5;
  float my_lp = 0.f;
  if (gw < EE) {
    int s = src[gw], t = dst[gw];
    const float* hs  = h  + (size_t)s * 64;
    const float* ht  = h  + (size_t)t * 64;
    const float* hls = hl + (size_t)s * 64;
    const float* hlt = hl + (size_t)t * 64;
    float hs0 = hs[lane],  hs1 = hs[lane + 32];
    float ht0 = ht[lane],  ht1 = ht[lane + 32];
    float ls0 = hls[lane], ls1 = hls[lane + 32];
    float lt0 = hlt[lane], lt1 = hlt[lane + 32];
    float sa0 = s_attn[lane], sa1 = s_attn[lane + 32];
    float se  = ls0 * sa0 * lt0 + ls1 * sa1 * lt1;
    float d0  = hs0 - ht0, d1 = hs1 - ht1;
    float sdf = d0 * d0 + d1 * d1;
    float e0  = ls0 - lt0, e1 = ls1 - lt1;
    float sds = e0 * e0 + e1 * e1;
    float dot = ls0 * lt0 + ls1 * lt1;
#pragma unroll
    for (int sh = 16; sh >= 1; sh >>= 1) {
      se  += __shfl_xor(se, sh);
      sdf += __shfl_xor(sdf, sh);
      sds += __shfl_xor(sds, sh);
      dot += __shfl_xor(dot, sh);
    }
    if (lane == 0) {
      float a0 = aw[0], a1 = aw[1];
      float mx = fmaxf(a0, a1);
      float x0 = expf(a0 - mx), x1 = expf(a1 - mx);
      float wt0 = x0 / (x0 + x1), wt1 = x1 / (x0 + x1);
      float betaw = 2.f / (expf(-beta[0]) + 1.f);
      float ee = el[s] + er[t] + se;
      float dd = wt0 * sdf + wt1 * sds;
      float v  = expf(ee - betaw * dd) + 1e-9f;
      ew[gw] = v;
      atomicAdd(&deg_s[s], v);
      atomicAdd(&deg_d[t], v);
      my_lp = dot * weights[gw];
    }
  }
  if (lane == 0) lds_lp[wib] = my_lp;
  __syncthreads();
  if (threadIdx.x == 0) {
    float s = 0.f;
#pragma unroll
    for (int i = 0; i < 8; ++i) s += lds_lp[i];
    atomicAdd(lp, s);
  }
}

__global__ void edgew_kernel(const int* __restrict__ src, const int* __restrict__ dst,
                             const float* __restrict__ ew,
                             const float* __restrict__ deg_s,
                             const float* __restrict__ deg_d,
                             const float* __restrict__ theta, float* __restrict__ w) {
  int e = blockIdx.x * blockDim.x + threadIdx.x;
  if (e >= EE) return;
  float perm = 1e-9f / (expf(-theta[0]) + 1.f);
  w[e] = rsqrtf(deg_s[src[e]]) * rsqrtf(deg_d[dst[e]]) * ew[e] + perm;
}

__global__ void zero_kernel(float* __restrict__ p, size_t n) {
  size_t i = (size_t)blockIdx.x * blockDim.x + threadIdx.x;
  size_t st = (size_t)gridDim.x * blockDim.x;
  for (; i < n; i += st) p[i] = 0.f;
}

__global__ void copy_kernel(const float* __restrict__ a, float* __restrict__ b, size_t n) {
  size_t i = (size_t)blockIdx.x * blockDim.x + threadIdx.x;
  size_t st = (size_t)gridDim.x * blockDim.x;
  for (; i < n; i += st) b[i] = a[i];
}

// agg[dst] += fp[src] * w  (wave per edge, 2 atomics per lane; L2-resident)
__global__ void scatter_kernel(const int* __restrict__ src, const int* __restrict__ dst,
                               const float* __restrict__ w, const float* __restrict__ fp,
                               float* __restrict__ agg) {
  int gw   = (blockIdx.x * blockDim.x + threadIdx.x) >> 5;
  int lane = threadIdx.x & 31;
  if (gw >= EE) return;
  int s = src[gw], t = dst[gw];
  float wv = w[gw];
  float f0 = fp[(size_t)s * 64 + lane] * wv;
  float f1 = fp[(size_t)s * 64 + 32 + lane] * wv;
  atomicAdd(&agg[(size_t)t * 64 + lane], f0);
  atomicAdd(&agg[(size_t)t * 64 + 32 + lane], f1);
}

// fp = (1-a)*agg + a*h, and re-zero agg for the next propagation round
__global__ void combine_kernel(float* __restrict__ agg, const float* __restrict__ h,
                               float* __restrict__ fp, size_t n) {
  size_t i = (size_t)blockIdx.x * blockDim.x + threadIdx.x;
  size_t st = (size_t)gridDim.x * blockDim.x;
  for (; i < n; i += st) {
    fp[i] = (1.f - ALPHA_) * agg[i] + ALPHA_ * h[i];
    agg[i] = 0.f;
  }
}

__global__ void logsoftmax_kernel(const float* __restrict__ fp, float* __restrict__ out) {
  int gw   = (blockIdx.x * blockDim.x + threadIdx.x) >> 5;
  int lane = threadIdx.x & 31;
  if (gw >= NN) return;
  float y0 = eluf(fp[(size_t)gw * 64 + lane]);
  float y1 = eluf(fp[(size_t)gw * 64 + 32 + lane]);
  float mx = fmaxf(y0, y1);
#pragma unroll
  for (int s = 16; s >= 1; s >>= 1) mx = fmaxf(mx, __shfl_xor(mx, s));
  float sm = expf(y0 - mx) + expf(y1 - mx);
#pragma unroll
  for (int s = 16; s >= 1; s >>= 1) sm += __shfl_xor(sm, s);
  float ls = logf(sm);
  out[(size_t)gw * 64 + lane]      = y0 - mx - ls;
  out[(size_t)gw * 64 + 32 + lane] = y1 - mx - ls;
}

__global__ void store_lp_kernel(const float* __restrict__ lp, float* __restrict__ out) {
  if (threadIdx.x == 0 && blockIdx.x == 0) out[(size_t)NN * 64] = lp[0];
}

// ---------------------------------------------------------------------------
extern "C" void kernel_launch(void* const* d_in, const int* in_sizes, int n_in,
                              void* d_out, int out_size, void* d_ws, size_t ws_size,
                              hipStream_t stream) {
  (void)in_sizes; (void)n_in; (void)out_size; (void)ws_size;
  const float* x       = (const float*)d_in[0];
  const int*   src     = (const int*)d_in[1];
  const int*   dst     = (const int*)d_in[2];
  const float* weights = (const float*)d_in[3];
  const float* latp    = (const float*)d_in[4];
  const float* W_in    = (const float*)d_in[5];
  const float* W_out   = (const float*)d_in[6];
  const float* Ws1     = (const float*)d_in[7];
  const float* b1      = (const float*)d_in[8];
  const float* Ws2     = (const float*)d_in[9];
  const float* b2      = (const float*)d_in[10];
  const float* attn_l  = (const float*)d_in[11];
  const float* attn_r  = (const float*)d_in[12];
  const float* s_attn  = (const float*)d_in[13];
  const float* beta    = (const float*)d_in[14];
  const float* theta   = (const float*)d_in[15];
  const float* aw      = (const float*)d_in[16];
  float* out = (float*)d_out;

  char* ws = (char*)d_ws;
  size_t o = 0;
  auto take = [&](size_t bytes) -> char* {
    char* p = ws + o;
    o += (bytes + 255) & ~(size_t)255;
    return p;
  };
  float*  h_buf  = (float*)take((size_t)NN * 64 * 4);
  float*  hl_buf = (float*)take((size_t)NN * 64 * 4);
  float*  fp_buf = (float*)take((size_t)NN * 64 * 4);
  float*  agg    = (float*)take((size_t)NN * 64 * 4);
  float*  el     = (float*)take((size_t)NN * 4);
  float*  er     = (float*)take((size_t)NN * 4);
  float*  deg_s  = (float*)take((size_t)NN * 4);
  float*  deg_d  = (float*)take((size_t)NN * 4);
  float*  ewb    = (float*)take((size_t)EE * 4);
  float*  wbuf   = (float*)take((size_t)EE * 4);
  float*  lp     = (float*)take(256);
  __bf16* winf   = (__bf16*)take((size_t)NFEAT_ * NHID_ * 2);
  __bf16* woutf  = (__bf16*)take((size_t)NHID_ * NCLS_ * 2);
  __bf16* ws1f   = (__bf16*)take((size_t)NCLS_ * NCLS_ * 2);
  __bf16* ws2f   = (__bf16*)take((size_t)NCLS_ * NCLS_ * 2);

  // Pack weights into WMMA B-fragment order (bf16); all L2-resident afterwards.
  frag_convert_kernel<<<(NFEAT_ * NHID_ + 255) / 256, 256, 0, stream>>>(W_in, winf, NFEAT_, NHID_);
  frag_convert_kernel<<<(NHID_ * NCLS_ + 255) / 256, 256, 0, stream>>>(W_out, woutf, NHID_, NCLS_);
  frag_convert_kernel<<<(NCLS_ * NCLS_ + 255) / 256, 256, 0, stream>>>(Ws1, ws1f, NCLS_, NCLS_);
  frag_convert_kernel<<<(NCLS_ * NCLS_ + 255) / 256, 256, 0, stream>>>(Ws2, ws2f, NCLS_, NCLS_);

  zero_kernel<<<1024, 256, 0, stream>>>(deg_s, (size_t)NN);
  zero_kernel<<<1024, 256, 0, stream>>>(deg_d, (size_t)NN);
  zero_kernel<<<1, 64, 0, stream>>>(lp, 64);
  zero_kernel<<<2048, 256, 0, stream>>>(agg, (size_t)NN * 64);

  const int nwaves = NN / 16;  // 6250 exact 16-row tiles
  gemm_h_kernel<<<(nwaves + 3) / 4, 128, 0, stream>>>(x, winf, woutf, h_buf);
  gemm_hl_kernel<<<(nwaves + 3) / 4, 128, 0, stream>>>(latp, ws1f, ws2f, b1, b2, hl_buf);

  attn_kernel<<<NN * 32 / 256, 256, 0, stream>>>(h_buf, attn_l, attn_r, el, er);

  edge_kernel<<<EE * 32 / 256, 256, 0, stream>>>(src, dst, h_buf, hl_buf, el, er,
                                                 s_attn, weights, beta, aw, ewb,
                                                 deg_s, deg_d, lp);

  edgew_kernel<<<(EE + 255) / 256, 256, 0, stream>>>(src, dst, ewb, deg_s, deg_d, theta, wbuf);

  copy_kernel<<<2048, 256, 0, stream>>>(h_buf, fp_buf, (size_t)NN * 64);

  for (int it = 0; it < KITER_; ++it) {
    scatter_kernel<<<EE * 32 / 256, 256, 0, stream>>>(src, dst, wbuf, fp_buf, agg);
    combine_kernel<<<2048, 256, 0, stream>>>(agg, h_buf, fp_buf, (size_t)NN * 64);
  }

  logsoftmax_kernel<<<NN * 32 / 256, 256, 0, stream>>>(fp_buf, out);
  store_lp_kernel<<<1, 64, 0, stream>>>(lp, out);
}
